// LatentGraphModule_55078660604306
// MI455X (gfx1250) — matrix operand order repaired and checked by reading, not compiled, not run on hardware
//
#include <hip/hip_runtime.h>
#include <hip/hip_bf16.h>

typedef __attribute__((ext_vector_type(16))) _Float16 v16h;
typedef __attribute__((ext_vector_type(8)))  _Float16 v8h;
typedef __attribute__((ext_vector_type(8)))  float    v8f;
typedef __attribute__((ext_vector_type(4)))  float    v4f;

#define B_DIM 64
#define N_DIM 512
#define D_DIM 256
#define NEGV  (-1000000000.0f)

#if defined(__has_builtin) && __has_builtin(__builtin_amdgcn_sched_barrier)
#define SCHED_FENCE() __builtin_amdgcn_sched_barrier(0)
#else
#define SCHED_FENCE()
#endif

__device__ __forceinline__ void top4_insert(float v, float& t0, float& t1,
                                            float& t2, float& t3) {
  if (v > t3) {
    if (v > t0)      { t3 = t2; t2 = t1; t1 = t0; t0 = v; }
    else if (v > t1) { t3 = t2; t2 = t1; t1 = v; }
    else if (v > t2) { t3 = t2; t2 = v; }
    else             { t3 = v; }
  }
}

// ---------------------------------------------------------------------------
// Kernel 0: convert Wq / Wk (f32, [256,256] row-major, out x in) to f16.
// ---------------------------------------------------------------------------
__global__ __launch_bounds__(256) void cvt_w_kernel(const float* __restrict__ Wq,
                                                    const float* __restrict__ Wk,
                                                    _Float16* __restrict__ Wqh,
                                                    _Float16* __restrict__ Wkh) {
  int i = blockIdx.x * 256 + threadIdx.x;   // 65536 elements
  Wqh[i] = (_Float16)Wq[i];
  Wkh[i] = (_Float16)Wk[i];
}

// ---------------------------------------------------------------------------
// Kernel 1: Q = X * Wq^T, K = X * Wk^T  (per batch: [512x256] = [512x256]x[256x256])
// Grid: B * (N/16) blocks, 128 threads (4 waves).
//   wave0: Q cols 0..127   wave1: Q cols 128..255
//   wave2: K cols 0..127   wave3: K cols 128..255
// Double-buffered over k-steps: loads for step ks+1 are issued (and fenced)
// before the WMMAs of step ks, so matrix math overlaps L2 latency.
// ---------------------------------------------------------------------------
__global__ __launch_bounds__(128) void proj_kernel(const float* __restrict__ X,
                                                   const _Float16* __restrict__ Wqh,
                                                   const _Float16* __restrict__ Wkh,
                                                   _Float16* __restrict__ Qh,
                                                   _Float16* __restrict__ Kh) {
  const int b     = blockIdx.x >> 5;       // 32 row tiles per batch
  const int r0    = (blockIdx.x & 31) * 16;
  const int tid   = threadIdx.x;
  const int wave  = tid >> 5;
  const int lane  = tid & 31;
  const int lhalf = lane >> 4;             // 0 or 1
  const int l15   = lane & 15;

  const _Float16* Wh = (wave >= 2) ? Wkh : Wqh;
  _Float16*       Oh = (wave >= 2) ? Kh  : Qh;
  const int ecb = (wave & 1) * 128;

  const float*    Xrow  = X + ((size_t)b * N_DIM + r0 + l15) * D_DIM + lhalf * 8;
  const _Float16* Wlane = Wh + (size_t)(ecb + l15) * D_DIM + lhalf * 8;

  auto loadA = [&](int ks) -> v16h {
    const int kb = ks * 32;
    const v4f x0 = __builtin_nontemporal_load((const v4f*)(Xrow + kb));
    const v4f x1 = __builtin_nontemporal_load((const v4f*)(Xrow + kb + 4));
    const v4f x2 = __builtin_nontemporal_load((const v4f*)(Xrow + kb + 16));
    const v4f x3 = __builtin_nontemporal_load((const v4f*)(Xrow + kb + 20));
    v16h a;
#pragma unroll
    for (int i = 0; i < 4; ++i) {
      a[i]      = (_Float16)x0[i];
      a[4 + i]  = (_Float16)x1[i];
      a[8 + i]  = (_Float16)x2[i];
      a[12 + i] = (_Float16)x3[i];
    }
    return a;
  };
  auto loadB = [&](int ct, int ks) -> v16h {
    const _Float16* p = Wlane + (size_t)ct * 16 * D_DIM + ks * 32;
    v8h lo = *(const v8h*)(p);
    v8h hi = *(const v8h*)(p + 16);
    v16h r;
#pragma unroll
    for (int i = 0; i < 8; ++i) { r[i] = lo[i]; r[8 + i] = hi[i]; }
    return r;
  };

  v8f acc[8] = {};
  v16h a = loadA(0);
  v16h bt[8];
#pragma unroll
  for (int ct = 0; ct < 8; ++ct) bt[ct] = loadB(ct, 0);

#pragma unroll
  for (int ks = 0; ks < 8; ++ks) {
    v16h an;
    v16h bn[8];
    if (ks < 7) {
      an = loadA(ks + 1);
#pragma unroll
      for (int ct = 0; ct < 8; ++ct) bn[ct] = loadB(ct, ks + 1);
    }
    SCHED_FENCE();   // keep next-step loads above this step's WMMAs
#pragma unroll
    for (int ct = 0; ct < 8; ++ct) {
      acc[ct] = __builtin_amdgcn_wmma_f32_16x16x32_f16(
          false, a, false, bt[ct], (short)0, acc[ct], false, false);
    }
    SCHED_FENCE();
    if (ks < 7) {
      a = an;
#pragma unroll
      for (int ct = 0; ct < 8; ++ct) bt[ct] = bn[ct];
    }
  }

  // C/D layout: VGPR j -> M = j + 8*lhalf, N = l15.  Regular-temporal stores:
  // Qh/Kh are re-read by attn_kernel and should stay resident in L2.
#pragma unroll
  for (int ct = 0; ct < 8; ++ct) {
    const int ecol = ecb + ct * 16 + l15;
#pragma unroll
    for (int j = 0; j < 8; ++j) {
      const int grow = r0 + j + lhalf * 8;
      Oh[((size_t)b * N_DIM + grow) * D_DIM + ecol] = (_Float16)acc[ct][j];
    }
  }
}

// ---------------------------------------------------------------------------
// Kernel 2: S = Q*K^T*scale, mask, top-4 threshold, softmax, * row-mask.
// Grid: B * (N/16) blocks, 128 threads (4 waves), wave w -> cols w*128..+127.
// ---------------------------------------------------------------------------
__global__ __launch_bounds__(128) void attn_kernel(const _Float16* __restrict__ Qh,
                                                   const _Float16* __restrict__ Kh,
                                                   const int* __restrict__ mask,
                                                   float* __restrict__ out) {
  __shared__ float sc[16][N_DIM];        // 32 KB score tile
  __shared__ int   cm[N_DIM];            // column masks
  __shared__ int   rm[16];               // row masks
  __shared__ float cand[16][8][4];       // per-row top-4 candidates
  __shared__ float thrS[16];
  __shared__ float rmaxS[16];
  __shared__ float psum[16][8];

  const int b     = blockIdx.x >> 5;
  const int r0    = (blockIdx.x & 31) * 16;
  const int tid   = threadIdx.x;
  const int wave  = tid >> 5;
  const int lane  = tid & 31;
  const int lhalf = lane >> 4;
  const int l15   = lane & 15;
  const float scale = 0.0625f;           // 256^-0.5

  if (tid < 16) rm[tid] = mask[b * N_DIM + r0 + tid];
  for (int c = tid; c < N_DIM; c += 128) cm[c] = mask[b * N_DIM + c];

  // ---- GEMM: 16x128 strip of S per wave, double-buffered over k ---------
  const _Float16* Qrow  = Qh + ((size_t)b * N_DIM + r0 + l15) * D_DIM + lhalf * 8;
  const _Float16* Klane = Kh + ((size_t)b * N_DIM + wave * 128 + l15) * D_DIM + lhalf * 8;

  auto loadA = [&](int ks) -> v16h {
    const int kb = ks * 32;
    v8h lo = *(const v8h*)(Qrow + kb);
    v8h hi = *(const v8h*)(Qrow + kb + 16);
    v16h a;
#pragma unroll
    for (int i = 0; i < 8; ++i) { a[i] = lo[i]; a[8 + i] = hi[i]; }
    return a;
  };
  auto loadB = [&](int ct, int ks) -> v16h {
    const _Float16* p = Klane + (size_t)ct * 16 * D_DIM + ks * 32;
    v8h lo = *(const v8h*)(p);
    v8h hi = *(const v8h*)(p + 16);
    v16h r;
#pragma unroll
    for (int i = 0; i < 8; ++i) { r[i] = lo[i]; r[8 + i] = hi[i]; }
    return r;
  };

  v8f acc[8] = {};
  v16h a = loadA(0);
  v16h bt[8];
#pragma unroll
  for (int ct = 0; ct < 8; ++ct) bt[ct] = loadB(ct, 0);

#pragma unroll
  for (int ks = 0; ks < 8; ++ks) {
    v16h an;
    v16h bn[8];
    if (ks < 7) {
      an = loadA(ks + 1);
#pragma unroll
      for (int ct = 0; ct < 8; ++ct) bn[ct] = loadB(ct, ks + 1);
    }
    SCHED_FENCE();   // keep next-step loads above this step's WMMAs
#pragma unroll
    for (int ct = 0; ct < 8; ++ct) {
      acc[ct] = __builtin_amdgcn_wmma_f32_16x16x32_f16(
          false, a, false, bt[ct], (short)0, acc[ct], false, false);
    }
    SCHED_FENCE();
    if (ks < 7) {
      a = an;
#pragma unroll
      for (int ct = 0; ct < 8; ++ct) bt[ct] = bn[ct];
    }
  }

  __syncthreads();   // masks visible; LDS tile free

  // ---- scale + mask, spill to LDS --------------------------------------
#pragma unroll
  for (int ct = 0; ct < 8; ++ct) {
    const int col  = wave * 128 + ct * 16 + l15;
    const bool cbad = (cm[col] == 0);
#pragma unroll
    for (int j = 0; j < 8; ++j) {
      const int m    = j + lhalf * 8;
      const int grow = r0 + m;
      const float s  = acc[ct][j] * scale;
      const bool bad = cbad || (rm[m] == 0) || (col == grow);
      sc[m][col] = bad ? NEGV : s;
    }
  }
  __syncthreads();

  // ---- top-4 per row: 8 threads/row, 64 strided cols each --------------
  const int row = tid >> 3;
  const int sub = tid & 7;
  {
    float t0 = -3.0e38f, t1 = -3.0e38f, t2 = -3.0e38f, t3 = -3.0e38f;
    for (int i = 0; i < 64; ++i) {
      top4_insert(sc[row][sub + i * 8], t0, t1, t2, t3);
    }
    cand[row][sub][0] = t0; cand[row][sub][1] = t1;
    cand[row][sub][2] = t2; cand[row][sub][3] = t3;
  }
  __syncthreads();

  if (sub == 0) {
    float t0 = -3.0e38f, t1 = -3.0e38f, t2 = -3.0e38f, t3 = -3.0e38f;
    for (int s = 0; s < 8; ++s)
      for (int q = 0; q < 4; ++q)
        top4_insert(cand[row][s][q], t0, t1, t2, t3);
    thrS[row]  = t3;   // 4th-largest value = threshold
    rmaxS[row] = t0;   // row max for stable softmax
  }
  __syncthreads();

  // ---- thresholded exp + row sum ---------------------------------------
  const float th = thrS[row];
  const float mx = rmaxS[row];
  float lsum = 0.0f;
  for (int i = 0; i < 64; ++i) {
    const int c = sub + i * 8;
    float v = sc[row][c];
    if (v < th) v = NEGV;          // keep ties with threshold, like reference
    const float e = __expf(v - mx);
    sc[row][c] = e;
    lsum += e;
  }
  psum[row][sub] = lsum;
  __syncthreads();

  float tot = 0.0f;
#pragma unroll
  for (int s = 0; s < 8; ++s) tot += psum[row][s];
  const float f = (float)rm[row] / tot;   // softmax * row mask

  // Streaming output: 67 MB written once, never re-read -> non-temporal.
  float* orow = out + ((size_t)b * N_DIM + r0 + row) * N_DIM;
  for (int i = 0; i < 64; ++i) {
    const int c = sub + i * 8;
    __builtin_nontemporal_store(sc[row][c] * f, orow + c);
  }
}

// ---------------------------------------------------------------------------
extern "C" void kernel_launch(void* const* d_in, const int* in_sizes, int n_in,
                              void* d_out, int out_size, void* d_ws, size_t ws_size,
                              hipStream_t stream) {
  (void)in_sizes; (void)n_in; (void)out_size; (void)ws_size;

  const float* X    = (const float*)d_in[0];   // [64,512,256] f32
  const int*   msk  = (const int*)  d_in[1];   // [64,512] i32
  const float* Wq   = (const float*)d_in[2];   // [256,256] f32
  const float* Wk   = (const float*)d_in[3];   // [256,256] f32
  float*       out  = (float*)d_out;           // [64,512,512] f32

  const size_t qkElems = (size_t)B_DIM * N_DIM * D_DIM;   // 8,388,608
  _Float16* Qh  = (_Float16*)d_ws;
  _Float16* Kh  = Qh  + qkElems;
  _Float16* Wqh = Kh  + qkElems;
  _Float16* Wkh = Wqh + (size_t)D_DIM * D_DIM;
  // ws usage: 2*16.78MB + 2*128KB ~= 33.8 MB

  cvt_w_kernel<<<(D_DIM * D_DIM) / 256, 256, 0, stream>>>(Wq, Wk, Wqh, Wkh);

  const int blocks = B_DIM * (N_DIM / 16);     // 2048
  proj_kernel<<<blocks, 128, 0, stream>>>(X, Wqh, Wkh, Qh, Kh);
  attn_kernel<<<blocks, 128, 0, stream>>>(Qh, Kh, msk, out);
}
